// BiMambaEncoderLayer_17626545783583
// MI455X (gfx1250) — matrix operand order repaired and checked by reading, not compiled, not run on hardware
//
#include <hip/hip_runtime.h>
#include <hip/hip_bf16.h>
#include <math.h>

// ---------------- problem constants ----------------
#define MTOT   16384      // BATCH * SEQ
#define BATCH  8
#define SEQ    2048
#define DMODEL 512
#define DINNER 512
#define NSTATE 32
#define DTRANK 32
#define DFF    2048

// ---------------- WMMA types ----------------
typedef __attribute__((ext_vector_type(16))) __bf16 v16bf;
typedef __attribute__((ext_vector_type(8)))  __bf16 v8bf;
typedef __attribute__((ext_vector_type(8)))  float  v8f;

// gfx1250 has native bf16 converts; let the compiler pick the packed cvt.
__device__ __forceinline__ __bf16 f2bf(float x) { return (__bf16)x; }

__device__ __forceinline__ float sigmoidf_(float x) { return 1.0f / (1.0f + __expf(-x)); }

// ---------------- tiled WMMA GEMM ----------------
// C[M,N] = act(A[M,K] * W[N,K]^T + bias) (+ C if accum)
// A row-major (lda), W row-major (ldw), C row-major (ldc).
// M multiple of 128, K multiple of 32; N arbitrary (guarded).
#define BM 128
#define BN 128
#define BK 32
#define LDT 40   // LDS row stride in bf16 (80B, 16B aligned, conflict-reducing)

__global__ __launch_bounds__(256)
void gemm_wmma_bf16(const float* __restrict__ A, int lda,
                    const float* __restrict__ W, int ldw,
                    float* __restrict__ C, int ldc,
                    int N, int K,
                    const float* __restrict__ bias, int act, int accum)
{
  __shared__ __bf16 aS[BM * LDT];
  __shared__ __bf16 wS[BN * LDT];

  const int tid  = threadIdx.x;
  const int lane = tid & 31;
  const int wave = tid >> 5;            // 0..7 -> 16-row strip
  const int bm   = blockIdx.y * BM;
  const int bn   = blockIdx.x * BN;
  const int hi   = (lane >= 16);        // lane half

  v8f acc[8] = {};

  for (int k0 = 0; k0 < K; k0 += BK) {
    // ---- prefetch next k-tile into cache (overlaps with this tile's WMMA) ----
    if (k0 + BK < K) {
      const int pr = tid >> 1;                 // 0..127
      const int pc = (tid & 1) << 4;           // 0 or 16 floats (64B halves)
      __builtin_prefetch(A + (size_t)(bm + pr) * lda + k0 + BK + pc, 0, 1);
      if (bn + pr < N)
        __builtin_prefetch(W + (size_t)(bn + pr) * ldw + k0 + BK + pc, 0, 1);
    }
    // ---- stage A tile (128x32 f32 -> bf16) ----
#pragma unroll
    for (int i = 0; i < 4; ++i) {
      int idx = tid + i * 256;          // 0..1023, 8 float4 per row
      int r = idx >> 3;
      int c = (idx & 7) << 2;
      const float4 v = *(const float4*)(A + (size_t)(bm + r) * lda + k0 + c);
      __bf16* d = &aS[r * LDT + c];
      d[0] = f2bf(v.x); d[1] = f2bf(v.y); d[2] = f2bf(v.z); d[3] = f2bf(v.w);
    }
    // ---- stage W tile (128x32, zero-fill rows >= N) ----
#pragma unroll
    for (int i = 0; i < 4; ++i) {
      int idx = tid + i * 256;
      int r = idx >> 3;
      int c = (idx & 7) << 2;
      float4 v = make_float4(0.f, 0.f, 0.f, 0.f);
      if (bn + r < N) v = *(const float4*)(W + (size_t)(bn + r) * ldw + k0 + c);
      __bf16* d = &wS[r * LDT + c];
      d[0] = f2bf(v.x); d[1] = f2bf(v.y); d[2] = f2bf(v.z); d[3] = f2bf(v.w);
    }
    __syncthreads();

    // A fragment: lanes 0-15 hold K{0..7,16..23}, lanes 16-31 K{8..15,24..31}
    const int ar = wave * 16 + (lane & 15);
    const int ka = hi ? 8 : 0;
    v8bf a0 = *(const v8bf*)(aS + ar * LDT + ka);
    v8bf a1 = *(const v8bf*)(aS + ar * LDT + ka + 16);
    v16bf afrag = __builtin_shufflevector(a0, a1,
        0,1,2,3,4,5,6,7,8,9,10,11,12,13,14,15);

#pragma unroll
    for (int nt = 0; nt < 8; ++nt) {
      // B fragment: lanes 0-15 hold K{0..7,16..23}, lanes 16-31 K{8..15,24..31}
      const int br = nt * 16 + (lane & 15);
      const int kb = hi ? 8 : 0;
      v8bf b0 = *(const v8bf*)(wS + br * LDT + kb);
      v8bf b1 = *(const v8bf*)(wS + br * LDT + kb + 16);
      v16bf bfrag = __builtin_shufflevector(b0, b1,
          0,1,2,3,4,5,6,7,8,9,10,11,12,13,14,15);
      acc[nt] = __builtin_amdgcn_wmma_f32_16x16x32_bf16(
          false, afrag, false, bfrag, (short)0, acc[nt], false, false);
    }
    __syncthreads();
  }

  // ---- epilogue: D VGPR e -> row (hi?8:0)+e, col = lane&15 ----
  const int rbase = bm + wave * 16 + (hi ? 8 : 0);
#pragma unroll
  for (int nt = 0; nt < 8; ++nt) {
    const int col = bn + nt * 16 + (lane & 15);
    if (col < N) {
      const float bv = bias ? bias[col] : 0.0f;
#pragma unroll
      for (int e = 0; e < 8; ++e) {
        float v = acc[nt][e] + bv;
        if (act == 1) v = 0.5f * v * (1.0f + erff(v * 0.70710678118654752f));
        float* cp = C + (size_t)(rbase + e) * ldc + col;
        if (accum) v += *cp;
        *cp = v;
      }
    }
  }
}

// ---------------- depthwise conv (k=2, causal in scan order) + SiLU ----------------
// xz: [MTOT,1024] (xs = cols 0..511), xc out: [MTOT,512]
__global__ __launch_bounds__(256)
void conv_silu_kernel(const float* __restrict__ xz, const float* __restrict__ cw,
                      const float* __restrict__ cb, float* __restrict__ xc, int dir)
{
  int i = blockIdx.x * blockDim.x + threadIdx.x;
  if (i >= MTOT * DINNER) return;
  int d = i & (DINNER - 1);
  int row = i >> 9;
  int t = row & (SEQ - 1);
  float prev = 0.f;
  if (dir == 0) { if (t > 0)       prev = xz[(size_t)(row - 1) * 1024 + d]; }
  else          { if (t < SEQ - 1) prev = xz[(size_t)(row + 1) * 1024 + d]; }
  float cur = xz[(size_t)row * 1024 + d];
  float v = prev * cw[d * 2 + 0] + cur * cw[d * 2 + 1] + cb[d];
  xc[i] = v * sigmoidf_(v);
}

// ---------------- softplus(dt + dt_b) in place ----------------
__global__ __launch_bounds__(256)
void softplus_kernel(float* __restrict__ dt, const float* __restrict__ dtb)
{
  int i = blockIdx.x * blockDim.x + threadIdx.x;
  if (i >= MTOT * DINNER) return;
  float x = dt[i] + dtb[i & (DINNER - 1)];
  dt[i] = (x > 20.f) ? x : log1pf(__expf(x));
}

// ---------------- selective scan: wave per (b,d), lane = state ----------------
__global__ __launch_bounds__(256)
void scan_kernel(const float* __restrict__ dt, const float* __restrict__ xc,
                 const float* __restrict__ dbl, const float* __restrict__ A_log,
                 float* __restrict__ y, int dir)
{
  int gw   = (blockIdx.x * blockDim.x + threadIdx.x) >> 5;  // 0..4095
  int lane = threadIdx.x & 31;
  int d = gw & (DINNER - 1);
  int b = gw >> 9;
  const float a = -__expf(A_log[d * NSTATE + lane]);
  float h = 0.f;
  const size_t rowbase = (size_t)b * SEQ;
  for (int s = 0; s < SEQ; ++s) {
    int t = dir ? (SEQ - 1 - s) : s;
    size_t r = rowbase + t;
    float dtv = dt[r * DINNER + d];
    float xv  = xc[r * DINNER + d];
    float Bv  = dbl[r * 96 + DTRANK + lane];
    float Cv  = dbl[r * 96 + DTRANK + NSTATE + lane];
    h = __expf(dtv * a) * h + (dtv * xv) * Bv;
    float p = h * Cv;
#pragma unroll
    for (int off = 16; off > 0; off >>= 1) p += __shfl_xor(p, off, 32);
    if (lane == 0) y[r * DINNER + d] = p;
  }
}

// ---------------- gate: y = (y + xc*Dp) * silu(z) ----------------
__global__ __launch_bounds__(256)
void gate_kernel(float* __restrict__ y, const float* __restrict__ xc,
                 const float* __restrict__ xz, const float* __restrict__ Dp)
{
  int i = blockIdx.x * blockDim.x + threadIdx.x;
  if (i >= MTOT * DINNER) return;
  int d = i & (DINNER - 1);
  int row = i >> 9;
  float z = xz[(size_t)row * 1024 + 512 + d];
  y[i] = (y[i] + xc[i] * Dp[d]) * (z * sigmoidf_(z));
}

// ---------------- out = LayerNorm(a + b), row width 512 ----------------
__global__ __launch_bounds__(256)
void add_ln_kernel(const float* __restrict__ a, const float* __restrict__ b,
                   const float* __restrict__ g, const float* __restrict__ be,
                   float* __restrict__ out)
{
  __shared__ float red[10];
  const size_t row = blockIdx.x;
  const int tid = threadIdx.x;
  const int lane = tid & 31, wave = tid >> 5;
  float v0 = a[row * 512 + tid]       + b[row * 512 + tid];
  float v1 = a[row * 512 + 256 + tid] + b[row * 512 + 256 + tid];
  float s = v0 + v1;
#pragma unroll
  for (int off = 16; off > 0; off >>= 1) s += __shfl_xor(s, off, 32);
  if (lane == 0) red[wave] = s;
  __syncthreads();
  if (tid == 0) { float t = 0.f; for (int i = 0; i < 8; ++i) t += red[i]; red[8] = t * (1.0f / 512.0f); }
  __syncthreads();
  const float mean = red[8];
  float d0 = v0 - mean, d1 = v1 - mean;
  float sq = d0 * d0 + d1 * d1;
#pragma unroll
  for (int off = 16; off > 0; off >>= 1) sq += __shfl_xor(sq, off, 32);
  if (lane == 0) red[wave] = sq;
  __syncthreads();
  if (tid == 0) { float t = 0.f; for (int i = 0; i < 8; ++i) t += red[i];
                  red[9] = rsqrtf(t * (1.0f / 512.0f) + 1e-5f); }
  __syncthreads();
  const float rs = red[9];
  out[row * 512 + tid]       = d0 * rs * g[tid]       + be[tid];
  out[row * 512 + 256 + tid] = d1 * rs * g[tid + 256] + be[tid + 256];
}

// ---------------- host orchestration ----------------
extern "C" void kernel_launch(void* const* d_in, const int* in_sizes, int n_in,
                              void* d_out, int out_size, void* d_ws, size_t ws_size,
                              hipStream_t stream)
{
  (void)in_sizes; (void)n_in; (void)out_size; (void)ws_size;
  const float* x = (const float*)d_in[0];
  // per-direction params: in_w, conv_w, conv_b, xproj_w, dt_w, dt_b, A_log, Dp, out_w
  const float* P[2][9];
  for (int dir = 0; dir < 2; ++dir)
    for (int j = 0; j < 9; ++j)
      P[dir][j] = (const float*)d_in[1 + dir * 9 + j];
  const float* c1w = (const float*)d_in[19];
  const float* c1b = (const float*)d_in[20];
  const float* c2w = (const float*)d_in[21];
  const float* c2b = (const float*)d_in[22];
  const float* n1g = (const float*)d_in[23];
  const float* n1b = (const float*)d_in[24];
  const float* n2g = (const float*)d_in[25];
  const float* n2b = (const float*)d_in[26];

  const size_t M = MTOT;
  float* ws  = (float*)d_ws;
  float* xz  = ws;               // M*1024 floats; part of FFN-hidden alias region
  float* xc  = ws + M * 1024;    // M*512 floats; part of FFN-hidden alias region
  float* dtb = ws + M * 1536;    // M*512 floats; part of FFN-hidden alias region
  float* hbf = ws;               // M*2048 floats (aliases xz/xc/dtb, used after mamba phase)
  float* dbl = ws + M * 2048;    // M*96 floats
  float* yb  = ws + M * 2144;    // M*512 floats
  float* nx  = ws + M * 2656;    // M*512 floats
  float* x1  = ws + M * 3168;    // M*512 floats

  const int EW_BLOCKS = (MTOT * DINNER) / 256;  // elementwise grid

  auto gemm = [&](const float* A, int lda, const float* W, int ldw,
                  float* C, int ldc, int N, int K,
                  const float* bias, int act, int accum) {
    dim3 grid((N + BN - 1) / BN, MTOT / BM);
    gemm_wmma_bf16<<<grid, 256, 0, stream>>>(A, lda, W, ldw, C, ldc, N, K, bias, act, accum);
  };

  for (int dir = 0; dir < 2; ++dir) {
    const float* in_w   = P[dir][0];
    const float* conv_w = P[dir][1];
    const float* conv_b = P[dir][2];
    const float* xpj_w  = P[dir][3];
    const float* dt_w   = P[dir][4];
    const float* dt_b   = P[dir][5];
    const float* A_log  = P[dir][6];
    const float* Dp     = P[dir][7];
    const float* out_w  = P[dir][8];

    // xz = x @ in_w^T                       [M,1024]
    gemm(x, DMODEL, in_w, DMODEL, xz, 1024, 1024, DMODEL, nullptr, 0, 0);
    // xc = silu(depthwise_conv(xs))         [M,512]
    conv_silu_kernel<<<EW_BLOCKS, 256, 0, stream>>>(xz, conv_w, conv_b, xc, dir);
    // dbl = xc @ xproj_w^T                  [M,96]
    gemm(xc, DINNER, xpj_w, DINNER, dbl, 96, 96, DINNER, nullptr, 0, 0);
    // dt = dbl[:, :32] @ dt_w^T             [M,512]
    gemm(dbl, 96, dt_w, DTRANK, dtb, DINNER, DINNER, DTRANK, nullptr, 0, 0);
    // dt = softplus(dt + dt_b)
    softplus_kernel<<<EW_BLOCKS, 256, 0, stream>>>(dtb, dt_b);
    // selective scan -> yb
    scan_kernel<<<(BATCH * DINNER * 32) / 256, 256, 0, stream>>>(dtb, xc, dbl, A_log, yb, dir);
    // yb = (yb + xc*Dp) * silu(z)
    gate_kernel<<<EW_BLOCKS, 256, 0, stream>>>(yb, xc, xz, Dp);
    // nx (+)= yb @ out_w^T                  [M,512]
    gemm(yb, DINNER, out_w, DINNER, nx, DMODEL, DMODEL, DINNER, nullptr, 0, dir);
  }

  // x1 = LN(x + nx)
  add_ln_kernel<<<MTOT, 256, 0, stream>>>(x, nx, n1g, n1b, x1);
  // h = gelu(x1 @ c1_w^T + c1_b)            [M,2048]  (aliases xz/xc/dtb)
  gemm(x1, DMODEL, c1w, DMODEL, hbf, DFF, DFF, DMODEL, c1b, 1, 0);
  // y2 = h @ c2_w^T + c2_b                  [M,512]
  gemm(hbf, DFF, c2w, DFF, yb, DMODEL, DMODEL, DFF, c2b, 0, 0);
  // out = LN(x1 + y2)
  add_ln_kernel<<<MTOT, 256, 0, stream>>>(x1, yb, n2g, n2b, (float*)d_out);
}
